// CausalMultiHeadSelfAttention_24404004176539
// MI455X (gfx1250) — matrix operand order repaired and checked by reading, not compiled, not run on hardware
//
#include <hip/hip_runtime.h>
#include <hip/hip_bf16.h>

#define BB 4
#define SS 2048
#define DMODEL 1024
#define HH 16
#define DKH 64

typedef __attribute__((ext_vector_type(16))) __bf16 v16bf;
typedef __attribute__((ext_vector_type(8)))  __bf16 v8bf;
typedef __attribute__((ext_vector_type(8)))  float  v8f;

// TDM descriptor group vector types (per probe: g0=uint32x4, g1=int32x8, ...)
typedef __attribute__((ext_vector_type(4))) unsigned int tdm_v4u;
typedef __attribute__((ext_vector_type(8))) int          tdm_v8i;
typedef __attribute__((ext_vector_type(4))) int          tdm_v4i;

// ROCm 7.2 (clang-22) declares the 5-arg builtin; amdgpu-toolchain
// (clang-23 + therock headers, which ship amd_gfx1250_TDM.h) the 6-arg form.
#if __has_include(<hip/amd_detail/amd_gfx1250_TDM.h>)
#define GFX1250_TDM_6ARG 1
#elif defined(__clang_major__) && (__clang_major__ >= 23)
#define GFX1250_TDM_6ARG 1
#endif

// ---------------------------------------------------------------------------
// TDM: async-copy a 2D tile (tile_d0 contiguous elems x tile_d1 rows) of a
// row-major (tensor_d1 x tensor_d0) bf16 tensor into LDS at lds_off.
// D# bit layout per CDNA5 ISA:
//   g0: [1:0]=count=1, [63:32]=lds_addr, [120:64]=global_addr, [127:126]=type=2
//   g1: [17:16]=data_size(1 -> 2B), [79:48]=tensor_dim0, [111:80]=tensor_dim1,
//       [127:112]=tile_dim0, [143:128]=tile_dim1, [207:160]=tensor_dim0_stride
// Groups 2/3 zero (<=2D tensor).
// ---------------------------------------------------------------------------
static __device__ __forceinline__ void tdm_load_tile_2d_bf16(
    unsigned lds_off, const void* gaddr, unsigned tensor_d0, unsigned tensor_d1,
    unsigned tile_d0, unsigned tile_d1, unsigned stride0_elems) {
  const unsigned long long ga = (unsigned long long)(uintptr_t)gaddr;
  tdm_v4u g0;
  g0[0] = 1u;                                               // count=1
  g0[1] = lds_off;                                          // lds_addr (bytes)
  g0[2] = (unsigned)(ga & 0xffffffffu);                     // global_addr[31:0]
  g0[3] = (unsigned)((ga >> 32) & 0x01ffffffu) | (2u << 30);// addr[56:32]|type=2
  tdm_v8i g1;
  g1[0] = (int)(1u << 16);                                  // data_size = 2 bytes
  g1[1] = (int)((tensor_d0 & 0xffffu) << 16);               // tensor_dim0[15:0]
  g1[2] = (int)(((tensor_d0 >> 16) & 0xffffu) |
                ((tensor_d1 & 0xffffu) << 16));             // dim0[31:16]|dim1[15:0]
  g1[3] = (int)(((tensor_d1 >> 16) & 0xffffu) |
                ((tile_d0 & 0xffffu) << 16));               // dim1[31:16]|tile_dim0
  g1[4] = (int)(tile_d1 & 0xffffu);                         // tile_dim1 (tile_dim2=0)
  g1[5] = (int)stride0_elems;                               // tensor_dim0_stride[31:0]
  g1[6] = 0;                                                // stride0[47:32]|dim1_stride lo
  g1[7] = 0;
  tdm_v4i z = {};
#if defined(GFX1250_TDM_6ARG)
  tdm_v8i z8 = {};
  __builtin_amdgcn_tensor_load_to_lds(g0, g1, z, z, z8, 0);
#else
  __builtin_amdgcn_tensor_load_to_lds(g0, g1, z, z, 0);
#endif
}

// ---------------------------------------------------------------------------
// Fragment loaders matching the CDNA5 wave32 16-bit WMMA VGPR layouts
// (CDNA5 ISA 7.12.2). Both assume K-contiguous storage.
// A (16x32, MxK): lanes 0-15 row m0+lane, K chunks [k0..k0+7] and [k0+16..k0+23]
//                 lanes 16-31 row m0+lane-16, chunks [k0+8..15] and [k0+24..31]
// ---------------------------------------------------------------------------
static __device__ __forceinline__ v16bf load_a16x32(const __bf16* base, int stride,
                                                    int m0, int k0, int lane) {
  const int r    = lane & 15;
  const int koff = (lane < 16) ? 0 : 8;
  const __bf16* p = base + (size_t)(m0 + r) * stride + (k0 + koff);
  v8bf lo = *(const v8bf*)p;
  v8bf hi = *(const v8bf*)(p + 16);
  v16bf out;
#pragma unroll
  for (int i = 0; i < 8; ++i) { out[i] = lo[i]; out[8 + i] = hi[i]; }
  return out;
}

// B (32x16, KxN) from an (N,K) row-major array ("NT" operand):
// lanes 0-15 col n0+lane, K [k0..k0+7] and [k0+16..k0+23]; lanes 16-31 col n0+lane-16, K [k0+8..15] and [k0+24..31]
static __device__ __forceinline__ v16bf load_b32x16(const __bf16* base, int stride,
                                                    int n0, int k0, int lane) {
  const int c    = lane & 15;
  const int koff = (lane < 16) ? 0 : 8;
  const __bf16* p = base + (size_t)(n0 + c) * stride + (k0 + koff);
  v8bf lo = *(const v8bf*)p;
  v8bf hi = *(const v8bf*)(p + 16);
  v16bf out;
#pragma unroll
  for (int i = 0; i < 8; ++i) { out[i] = lo[i]; out[8 + i] = hi[i]; }
  return out;
}

static __device__ __forceinline__ v8f wmma_bf16(v16bf a, v16bf b, v8f c) {
  return __builtin_amdgcn_wmma_f32_16x16x32_bf16(false, a, false, b, (short)0, c,
                                                 false, false);
}

// ---------------------------------------------------------------------------
// fp32 -> bf16 convert
// ---------------------------------------------------------------------------
__global__ void f32_to_bf16_kernel(const float* __restrict__ in,
                                   __bf16* __restrict__ out, int n) {
  int i = blockIdx.x * blockDim.x + threadIdx.x;
  if (i < n) out[i] = (__bf16)in[i];
}

// ---------------------------------------------------------------------------
// C[m,n] = sum_k A[m,k] * Bw[n,k]   (A: MxK row-major, Bw: NxK row-major)
// block = 128 threads (4 waves) computing a 64x64 tile. The shared 64-col
// B (weight) tile is staged into LDS by the Tensor Data Mover (wave 0 issues
// the descriptor, waits TENSORcnt, barrier releases the other waves), so the
// weight operand is fetched once per block instead of once per wave.
// ---------------------------------------------------------------------------
__global__ void __launch_bounds__(128)
gemm_nt_bf16_kernel(const __bf16* __restrict__ A, const __bf16* __restrict__ Bw,
                    float* __restrict__ C, int M, int N, int K) {
  const int lane = threadIdx.x & 31;
  const int m0 = blockIdx.y * 64 + (threadIdx.x >> 5) * 16;
  const int n0 = blockIdx.x * 64;

  __shared__ __align__(16) __bf16 Btile[64 * 32];  // 64 rows x 32 K, 4 KB

  v8f acc0 = {}, acc1 = {}, acc2 = {}, acc3 = {};
  for (int k0 = 0; k0 < K; k0 += 32) {
    if (threadIdx.x < 32) {  // wave 0 drives the TDM
      tdm_load_tile_2d_bf16((unsigned)(uintptr_t)Btile,
                            Bw + (size_t)n0 * K + k0,
                            (unsigned)K, (unsigned)N,
                            32u, 64u, (unsigned)K);
      __builtin_amdgcn_s_wait_tensorcnt(0);
    }
    v16bf a = load_a16x32(A, K, m0, k0, lane);  // overlap with TDM
    __syncthreads();                            // B tile visible to all waves
    v16bf b0 = load_b32x16(Btile, 32, 0,  0, lane);
    v16bf b1 = load_b32x16(Btile, 32, 16, 0, lane);
    v16bf b2 = load_b32x16(Btile, 32, 32, 0, lane);
    v16bf b3 = load_b32x16(Btile, 32, 48, 0, lane);
    acc0 = wmma_bf16(a, b0, acc0);
    acc1 = wmma_bf16(a, b1, acc1);
    acc2 = wmma_bf16(a, b2, acc2);
    acc3 = wmma_bf16(a, b3, acc3);
    __syncthreads();                            // reads done before next TDM
  }

  const int roff = (lane < 16) ? 0 : 8;
  const int col  = n0 + (lane & 15);
#pragma unroll
  for (int r = 0; r < 8; ++r) {
    size_t row = (size_t)(m0 + r + roff) * N;
    C[row + col]      = acc0[r];
    C[row + col + 16] = acc1[r];
    C[row + col + 32] = acc2[r];
    C[row + col + 48] = acc3[r];
  }
}

// ---------------------------------------------------------------------------
// RoPE + relayout: qraw/kraw/vraw are (B,S,DMODEL) fp32.
//   Q,K -> bf16 (B,H,S,DK) with rotary applied
//   V   -> bf16 (B,H,DK,S) (transposed so PV B-fragments are contiguous)
// one thread per (b,s,h,pair)
// ---------------------------------------------------------------------------
__global__ void rope_pack_kernel(const float* __restrict__ qraw,
                                 const float* __restrict__ kraw,
                                 const float* __restrict__ vraw,
                                 const int* __restrict__ pos,
                                 __bf16* __restrict__ Q, __bf16* __restrict__ K,
                                 __bf16* __restrict__ Vt) {
  int idx = blockIdx.x * blockDim.x + threadIdx.x;
  const int total = BB * SS * HH * 32;
  if (idx >= total) return;
  const int i = idx & 31;                 // pair index 0..31
  const int h = (idx >> 5) & (HH - 1);
  const int s = (idx >> 9) & (SS - 1);
  const int b = idx >> 20;
  const int e0 = 2 * i;

  const float inv_freq = __powf(10000.0f, -(float)e0 / (float)DKH);
  const float ang = (float)pos[s] * inv_freq;
  float cs, sn;
  __sincosf(ang, &sn, &cs);

  const size_t src = ((size_t)b * SS + s) * DMODEL + h * DKH;
  const size_t dst = (((size_t)b * HH + h) * SS + s) * DKH;

  float x1 = qraw[src + e0], x2 = qraw[src + e0 + 1];
  Q[dst + e0]     = (__bf16)(x1 * cs - x2 * sn);
  Q[dst + e0 + 1] = (__bf16)(x1 * sn + x2 * cs);

  x1 = kraw[src + e0]; x2 = kraw[src + e0 + 1];
  K[dst + e0]     = (__bf16)(x1 * cs - x2 * sn);
  K[dst + e0 + 1] = (__bf16)(x1 * sn + x2 * cs);

  const size_t vb = ((size_t)b * HH + h) * DKH;
  Vt[(vb + e0) * SS + s]     = (__bf16)vraw[src + e0];
  Vt[(vb + e0 + 1) * SS + s] = (__bf16)vraw[src + e0 + 1];
}

// ---------------------------------------------------------------------------
// Flash attention, causal. One wave (32 threads) per (b, h, 16-query tile).
// Streams 32 keys per iteration; scores = 4 WMMAs, PV = 4 WMMAs.
// Online softmax with width-16 shuffle reductions (C layout: row's 16 cols
// live across 16 lanes at the same accumulator register).
// O written fp32 as (B, S, DMODEL) with head offset h*DK.
// ---------------------------------------------------------------------------
__global__ void __launch_bounds__(32)
flash_attn_kernel(const __bf16* __restrict__ Q, const __bf16* __restrict__ Km,
                  const __bf16* __restrict__ Vt, float* __restrict__ O) {
  const int lane = threadIdx.x;
  const int qtiles = SS / 16;
  const int qt = blockIdx.x % qtiles;
  const int bh = blockIdx.x / qtiles;
  const int b = bh / HH, h = bh % HH;
  const int q0 = qt * 16;

  const __bf16* Qp = Q  + (size_t)bh * SS * DKH;
  const __bf16* Kp = Km + (size_t)bh * SS * DKH;
  const __bf16* Vp = Vt + (size_t)bh * DKH * SS;

  const v16bf qa0 = load_a16x32(Qp, DKH, q0, 0,  lane);
  const v16bf qa1 = load_a16x32(Qp, DKH, q0, 32, lane);

  v8f o0 = {}, o1 = {}, o2 = {}, o3 = {};
  float mrow[8], lrow[8];
#pragma unroll
  for (int r = 0; r < 8; ++r) { mrow[r] = -1e30f; lrow[r] = 0.0f; }

  const int roff = (lane < 16) ? 0 : 8;
  const int colk = lane & 15;
  const float scale = 0.125f;  // 1/sqrt(64)

  __shared__ __align__(16) __bf16 Plds[16 * 32];

  const int kend = q0 + 16;  // exclusive causal bound
  for (int k0 = 0; k0 < kend; k0 += 32) {
    // ---- scores: S[q, k0..k0+31] = Q (16xDK) * K^T -----------------------
    v8f s1 = {}, s2 = {};
    s1 = wmma_bf16(qa0, load_b32x16(Kp, DKH, k0,      0,  lane), s1);
    s1 = wmma_bf16(qa1, load_b32x16(Kp, DKH, k0,      32, lane), s1);
    s2 = wmma_bf16(qa0, load_b32x16(Kp, DKH, k0 + 16, 0,  lane), s2);
    s2 = wmma_bf16(qa1, load_b32x16(Kp, DKH, k0 + 16, 32, lane), s2);

    // ---- online softmax --------------------------------------------------
    float corr[8];
#pragma unroll
    for (int r = 0; r < 8; ++r) {
      const int q = q0 + r + roff;
      float a1 = (k0 + colk      <= q) ? s1[r] * scale : -1e30f;
      float a2 = (k0 + 16 + colk <= q) ? s2[r] * scale : -1e30f;
      float mx = fmaxf(a1, a2);
#pragma unroll
      for (int sft = 1; sft < 16; sft <<= 1)
        mx = fmaxf(mx, __shfl_xor(mx, sft, 32));
      const float mn = fmaxf(mrow[r], mx);
      const float p1 = __expf(a1 - mn);
      const float p2 = __expf(a2 - mn);
      float rs = p1 + p2;
#pragma unroll
      for (int sft = 1; sft < 16; sft <<= 1)
        rs += __shfl_xor(rs, sft, 32);
      const float cf = __expf(mrow[r] - mn);
      lrow[r] = lrow[r] * cf + rs;
      mrow[r] = mn;
      corr[r] = cf;
      s1[r] = p1;
      s2[r] = p2;
    }
#pragma unroll
    for (int r = 0; r < 8; ++r) {
      o0[r] *= corr[r]; o1[r] *= corr[r]; o2[r] *= corr[r]; o3[r] *= corr[r];
    }

    // ---- transpose P via LDS into an A fragment --------------------------
    __syncthreads();
#pragma unroll
    for (int r = 0; r < 8; ++r) {
      Plds[(r + roff) * 32 + colk]      = (__bf16)s1[r];
      Plds[(r + roff) * 32 + 16 + colk] = (__bf16)s2[r];
    }
    __syncthreads();
    const v16bf pa = load_a16x32((const __bf16*)Plds, 32, 0, 0, lane);

    // ---- O += P (16x32) * V (32xDK), Vt is (DK,S) so B-frags contiguous --
    o0 = wmma_bf16(pa, load_b32x16(Vp, SS, 0,  k0, lane), o0);
    o1 = wmma_bf16(pa, load_b32x16(Vp, SS, 16, k0, lane), o1);
    o2 = wmma_bf16(pa, load_b32x16(Vp, SS, 32, k0, lane), o2);
    o3 = wmma_bf16(pa, load_b32x16(Vp, SS, 48, k0, lane), o3);
  }

  float* Op = O + ((size_t)b * SS + q0) * DMODEL + h * DKH;
#pragma unroll
  for (int r = 0; r < 8; ++r) {
    const float inv = 1.0f / lrow[r];
    const size_t row = (size_t)(r + roff) * DMODEL;
    Op[row + colk]      = o0[r] * inv;
    Op[row + colk + 16] = o1[r] * inv;
    Op[row + colk + 32] = o2[r] * inv;
    Op[row + colk + 48] = o3[r] * inv;
  }
}

// ---------------------------------------------------------------------------
extern "C" void kernel_launch(void* const* d_in, const int* in_sizes, int n_in,
                              void* d_out, int out_size, void* d_ws, size_t ws_size,
                              hipStream_t stream) {
  (void)in_sizes; (void)n_in; (void)out_size; (void)ws_size;
  const float* x   = (const float*)d_in[0];
  const float* Wq  = (const float*)d_in[1];
  const float* Wk  = (const float*)d_in[2];
  const float* Wv  = (const float*)d_in[3];
  const float* Wo  = (const float*)d_in[4];
  const int*   pos = (const int*)d_in[5];

  const size_t NX = (size_t)BB * SS * DMODEL;   // 8,388,608 activation elems
  const size_t DD = (size_t)DMODEL * DMODEL;    // 1,048,576 weight elems

  char* base = (char*)d_ws;
  size_t off = 0;
  auto alloc = [&](size_t bytes) -> char* {
    char* p = base + off;
    off += (bytes + 255) & ~(size_t)255;
    return p;
  };

  __bf16* xbf  = (__bf16*)alloc(NX * 2);
  __bf16* wqb  = (__bf16*)alloc(DD * 2);
  __bf16* wkb  = (__bf16*)alloc(DD * 2);
  __bf16* wvb  = (__bf16*)alloc(DD * 2);
  __bf16* wob  = (__bf16*)alloc(DD * 2);
  float*  qraw = (float*)alloc(NX * 4);
  float*  kraw = (float*)alloc(NX * 4);
  float*  vraw = (float*)alloc(NX * 4);
  __bf16* Qbf  = (__bf16*)alloc(NX * 2);
  __bf16* Kbf  = (__bf16*)alloc(NX * 2);
  __bf16* Vtb  = (__bf16*)alloc(NX * 2);
  float*  attnO   = qraw;            // reuse: qraw dead after rope_pack
  __bf16* attnObf = (__bf16*)kraw;   // reuse: kraw dead after rope_pack

  // 1) bf16 conversions
  f32_to_bf16_kernel<<<(int)((NX + 255) / 256), 256, 0, stream>>>(x, xbf, (int)NX);
  f32_to_bf16_kernel<<<(int)((DD + 255) / 256), 256, 0, stream>>>(Wq, wqb, (int)DD);
  f32_to_bf16_kernel<<<(int)((DD + 255) / 256), 256, 0, stream>>>(Wk, wkb, (int)DD);
  f32_to_bf16_kernel<<<(int)((DD + 255) / 256), 256, 0, stream>>>(Wv, wvb, (int)DD);
  f32_to_bf16_kernel<<<(int)((DD + 255) / 256), 256, 0, stream>>>(Wo, wob, (int)DD);

  // 2) QKV projections: (B*S, D) x (D, D)^T
  dim3 gblk(128);
  dim3 ggrd(DMODEL / 64, (BB * SS) / 64);
  gemm_nt_bf16_kernel<<<ggrd, gblk, 0, stream>>>(xbf, wqb, qraw, BB * SS, DMODEL, DMODEL);
  gemm_nt_bf16_kernel<<<ggrd, gblk, 0, stream>>>(xbf, wkb, kraw, BB * SS, DMODEL, DMODEL);
  gemm_nt_bf16_kernel<<<ggrd, gblk, 0, stream>>>(xbf, wvb, vraw, BB * SS, DMODEL, DMODEL);

  // 3) RoPE + head relayout (+ V transpose)
  const size_t nrope = (size_t)BB * SS * HH * 32;
  rope_pack_kernel<<<(int)((nrope + 255) / 256), 256, 0, stream>>>(
      qraw, kraw, vraw, pos, Qbf, Kbf, Vtb);

  // 4) causal flash attention
  flash_attn_kernel<<<BB * HH * (SS / 16), 32, 0, stream>>>(Qbf, Kbf, Vtb, attnO);

  // 5) output projection into d_out
  f32_to_bf16_kernel<<<(int)((NX + 255) / 256), 256, 0, stream>>>(attnO, attnObf, (int)NX);
  gemm_nt_bf16_kernel<<<ggrd, gblk, 0, stream>>>(attnObf, wob, (float*)d_out,
                                                 BB * SS, DMODEL, DMODEL);
}
